// Task_36275293782743
// MI455X (gfx1250) — matrix operand order repaired and checked by reading, not compiled, not run on hardware
//
#include <hip/hip_runtime.h>
#include <math.h>

// ---------------- problem constants (from reference) ----------------
static constexpr int Bb    = 16;
static constexpr int Nn    = 4096;
static constexpr int D_IN  = 128;
static constexpr int D_OUT = 128;
static constexpr int HID   = 256;
static constexpr int E     = 8;
static constexpr int T     = Bb * Nn;   // 65536 tokens

// padded LDS row strides (elements) -> conflict-free ds_load_b128
static constexpr int W1_STRIDE = 136;   // 128 + 8  (272 B rows)
static constexpr int W2_STRIDE = 40;    // 32 + 8   (80 B rows)
static constexpr int H_STRIDE  = 40;    // 32 + 8

// ---------------- trivially-copyable vector types ----------------
typedef __attribute__((ext_vector_type(16))) __bf16   v16bf;
typedef __attribute__((ext_vector_type(8)))  float    v8f;
typedef __attribute__((ext_vector_type(4)))  unsigned u32x4;
typedef __attribute__((ext_vector_type(2)))  unsigned u32x2;
typedef __attribute__((ext_vector_type(4)))  float    f32x4;
typedef __attribute__((ext_vector_type(4)))  int      i32x4;
typedef __attribute__((ext_vector_type(8)))  int      i32x8;

union Frag32B {           // one WMMA bf16 operand: 16 bf16 = 32 bytes = 8 VGPRs
    u32x4 q[2];
    v16bf v;
};

__device__ __forceinline__ unsigned short f2bf(float f) {
    unsigned u = __float_as_uint(f);
    u += 0x7fffu + ((u >> 16) & 1u);
    return (unsigned short)(u >> 16);
}

#if __has_builtin(__builtin_amdgcn_tensor_load_to_lds) && \
    __has_builtin(__builtin_amdgcn_s_wait_tensorcnt)
#define HAVE_TDM 1
#else
#define HAVE_TDM 0
#endif

#if HAVE_TDM
// Issue one 2D TDM load (bf16 elements) with LDS row padding.
// This toolchain exposes the 6-arg builtin:
//   (uint32x4 g0, int32x8 g1, int32x4 g2, int32x4 g3, int32x8 g4, i32 cpol)
// Our tiles are <=2D, so groups 2/3 (and the trailing group) are zero.
// pad_icode: pad_interval code (interval = 2^(code+1) dwords)
// pad_acode: pad_amount code (amount = code+1 dwords)
__device__ __forceinline__ void tdm_load_2d(
    unsigned lds_off, unsigned long long gaddr,
    unsigned tensor_d0, unsigned tensor_d1,
    unsigned tile_d0, unsigned tile_d1,
    unsigned long long stride0,
    unsigned pad_icode, unsigned pad_acode)
{
    u32x4 g0; i32x8 g1; i32x4 g2; i32x4 g3; i32x8 g4;
    // ---- group 0: count=1, lds_addr, global_addr, type=2 ----
    g0[0] = 1u;                                        // count=1, user mode
    g0[1] = lds_off;                                   // LDS byte address
    g0[2] = (unsigned)(gaddr & 0xFFFFFFFFull);         // global_addr[31:0]
    g0[3] = (unsigned)((gaddr >> 32) & 0x01FFFFFFull)  // global_addr[56:32]
          | (2u << 30);                                // type=2 ("image")
    // ---- group 1 ----
    g1[0] = (int)((1u << 16)                           // data_size=1 (2 bytes)
          | (1u << 20)                                 // pad_enable
          | (pad_icode << 22) | (pad_acode << 25));
    g1[1] = (int)((tensor_d0 & 0xFFFFu) << 16);        // abar=0 | dim0[15:0]
    g1[2] = (int)((tensor_d0 >> 16) | ((tensor_d1 & 0xFFFFu) << 16));
    g1[3] = (int)((tensor_d1 >> 16) | (tile_d0 << 16));
    g1[4] = (int)(tile_d1 & 0xFFFFu);                  // tile_dim2 = 0
    g1[5] = (int)(unsigned)(stride0 & 0xFFFFFFFFull);  // dim0_stride[31:0]
    g1[6] = (int)((unsigned)(stride0 >> 32) & 0xFFFFu);// dim0_stride[47:32]
    g1[7] = 0;                                         // dim1_stride = 0
    g2[0] = g2[1] = g2[2] = g2[3] = 0;
    g3[0] = g3[1] = g3[2] = g3[3] = 0;
    g4[0] = g4[1] = g4[2] = g4[3] = 0;
    g4[4] = g4[5] = g4[6] = g4[7] = 0;
    __builtin_amdgcn_tensor_load_to_lds(g0, g1, g2, g3, g4, 0);
}
#endif

// ---------------- fp32 -> bf16 conversion (x4 vectorized) ----------------
__global__ __launch_bounds__(256) void cvt_bf16_kernel(
    const float* __restrict__ in, unsigned short* __restrict__ out, int n4)
{
    int i = blockIdx.x * blockDim.x + threadIdx.x;
    if (i >= n4) return;
    f32x4 f = reinterpret_cast<const f32x4*>(in)[i];
    union { unsigned short s[4]; u32x2 u; } o;
    o.s[0] = f2bf(f[0]); o.s[1] = f2bf(f[1]); o.s[2] = f2bf(f[2]); o.s[3] = f2bf(f[3]);
    reinterpret_cast<u32x2*>(out)[i] = o.u;
}

// ---------------- gate path: SharedGate + residual + masked softmax ---------
__global__ __launch_bounds__(256) void gate_kernel(
    const float* __restrict__ x,      // [T, D_IN]
    const int*   __restrict__ act,    // [B, E]
    const float* __restrict__ sgw1,   // [16, D_IN]
    const float* __restrict__ sgw2,   // [D_IN, 16]
    const float* __restrict__ gatew,  // [E, D_IN]
    float*       __restrict__ gwout)  // [T, E]
{
    int t = blockIdx.x * 256 + threadIdx.x;
    if (t >= T) return;
    const float* xr = x + (size_t)t * D_IN;

    float s1[16];
    #pragma unroll
    for (int j = 0; j < 16; ++j) {
        float acc = 0.f;
        for (int k = 0; k < D_IN; ++k) acc = fmaf(xr[k], sgw1[j * D_IN + k], acc);
        s1[j] = acc > 0.f ? acc : 0.f;
    }

    float sc[E];
    #pragma unroll
    for (int e = 0; e < E; ++e) sc[e] = 0.f;
    for (int k = 0; k < D_IN; ++k) {
        float sh = xr[k];                         // residual
        #pragma unroll
        for (int j = 0; j < 16; ++j) sh = fmaf(s1[j], sgw2[k * 16 + j], sh);
        #pragma unroll
        for (int e = 0; e < E; ++e) sc[e] = fmaf(sh, gatew[e * D_IN + k], sc[e]);
    }

    int b = t >> 12;                              // t / N
    int m8[E];
    float mx = -INFINITY;
    #pragma unroll
    for (int e = 0; e < E; ++e) {
        m8[e] = act[b * E + e];
        if (m8[e] && sc[e] > mx) mx = sc[e];
    }
    float sum = 0.f, wv[E];
    #pragma unroll
    for (int e = 0; e < E; ++e) {
        wv[e] = m8[e] ? __expf(sc[e] - mx) : 0.f;
        sum += wv[e];
    }
    float inv = sum > 0.f ? 1.0f / sum : 0.f;
    #pragma unroll
    for (int e = 0; e < E; ++e) gwout[(size_t)t * E + e] = wv[e] * inv;
}

// ---------------- main: dense 8-expert MLP, WMMA bf16 + TDM staging ---------
// Block = 256 threads = 8 waves; each wave owns 16 tokens. Grid = T/128 = 512.
// Per step i = e*8+hc (64 steps): TDM double-buffers an 8 KB w1 chunk
// (HID rows [hc*32,hc*32+32) x D_IN) and an 8 KB w2 k-slice
// (D_OUT rows x HID cols [hc*32,hc*32+32)) into padded LDS while all waves
// run WMMAs against the previous chunk.
__global__ __launch_bounds__(256) void moe_expert_kernel(
    const unsigned short* __restrict__ xb,   // [T, D_IN]       bf16
    const unsigned short* __restrict__ w1b,  // [E, HID, D_IN]  bf16
    const unsigned short* __restrict__ w2b,  // [E, D_OUT, HID] bf16
    const float*          __restrict__ gw,   // [T, E]          fp32 gate weights
    float*                __restrict__ out)  // [T, D_OUT] (+ aux at T*D_OUT)
{
    __shared__ __attribute__((aligned(16))) unsigned short w1buf[2][32 * W1_STRIDE];
    __shared__ __attribute__((aligned(16))) unsigned short w2buf[2][128 * W2_STRIDE];
    __shared__ __attribute__((aligned(16))) unsigned short hbuf[8][16 * H_STRIDE];
    __shared__ float gsh[128 * E];

    const int tid  = threadIdx.x;
    const int wave = tid >> 5;
    const int lane = tid & 31;
    const int hf   = lane >> 4;     // lane half selects K sub-range (ISA 7.12.2)
    const int l16  = lane & 15;
    const int tok0 = blockIdx.x * 128;
    const int mrow = tok0 + wave * 16;

    for (int i = tid; i < 128 * E; i += 256) gsh[i] = gw[(size_t)tok0 * E + i];

    // A fragments of x (16x32 bf16 each), 4 K-chunks covering D_IN=128.
    Frag32B a[4];
    {
        const unsigned short* xr = xb + (size_t)(mrow + l16) * D_IN;
        #pragma unroll
        for (int kc = 0; kc < 4; ++kc) {
            const unsigned short* p = xr + kc * 32 + hf * 8;
            a[kc].q[0] = *reinterpret_cast<const u32x4*>(p);
            a[kc].q[1] = *reinterpret_cast<const u32x4*>(p + 16);
        }
    }

    v8f oacc[8];
    #pragma unroll
    for (int i = 0; i < 8; ++i)
        #pragma unroll
        for (int j = 0; j < 8; ++j) oacc[i][j] = 0.0f;

    unsigned short* hb = &hbuf[wave][0];

#if HAVE_TDM
    // prologue: DMA chunk 0 into buffer 0 (single issuing wave; EXEC ignored)
    if (wave == 0) {
        tdm_load_2d((unsigned)(size_t)&w1buf[0][0],
                    (unsigned long long)(size_t)w1b,          // e=0, hc=0
                    4096u, 1u, 4096u, 1u, 4096ull, 5u, 3u);   // 1D, pad 16B/256B
        tdm_load_2d((unsigned)(size_t)&w2buf[0][0],
                    (unsigned long long)(size_t)w2b,
                    32u, 128u, 32u, 128u, (unsigned long long)HID, 3u, 3u);
    }
#endif

    for (int i = 0; i < E * 8; ++i) {            // i = e*8 + hc
        const int e  = i >> 3;
        const int hc = i & 7;
        const int buf = i & 1;

#if HAVE_TDM
        if (wave == 0) __builtin_amdgcn_s_wait_tensorcnt(0);  // chunk i landed
        __syncthreads();                                      // visible to all
        if (wave == 0 && i + 1 < E * 8) {                     // prefetch i+1
            const int en  = (i + 1) >> 3;
            const int hcn = (i + 1) & 7;
            tdm_load_2d((unsigned)(size_t)&w1buf[buf ^ 1][0],
                        (unsigned long long)(size_t)(w1b + ((size_t)en * HID + (size_t)hcn * 32) * D_IN),
                        4096u, 1u, 4096u, 1u, 4096ull, 5u, 3u);
            tdm_load_2d((unsigned)(size_t)&w2buf[buf ^ 1][0],
                        (unsigned long long)(size_t)(w2b + (size_t)en * D_OUT * HID + (size_t)hcn * 32),
                        32u, 128u, 32u, 128u, (unsigned long long)HID, 3u, 3u);
        }
#else
        {   // fallback: cooperative staging with the same padded layouts
            const unsigned short* w1g = w1b + ((size_t)e * HID + (size_t)hc * 32) * D_IN;
            const unsigned short* w2g = w2b + (size_t)e * D_OUT * HID + (size_t)hc * 32;
            for (int idx = tid; idx < 32 * 128; idx += 256)
                w1buf[buf][(idx >> 7) * W1_STRIDE + (idx & 127)] = w1g[idx];
            for (int idx = tid; idx < 128 * 32; idx += 256)
                w2buf[buf][(idx >> 5) * W2_STRIDE + (idx & 31)] =
                    w2g[(size_t)(idx >> 5) * HID + (idx & 31)];
        }
        __syncthreads();
#endif

        const unsigned short* w1c = &w1buf[buf][0];   // [32][W1_STRIDE]
        const unsigned short* w2c = &w2buf[buf][0];   // [128][W2_STRIDE]

        // gate weight per C/D row (rows j + 8*hf of this wave's token tile)
        float wrow[8];
        #pragma unroll
        for (int j = 0; j < 8; ++j)
            wrow[j] = gsh[(wave * 16 + j + 8 * hf) * E + e];

        // -------- layer 1: two 16x16 h tiles for HID cols [hc*32, hc*32+32)
        #pragma unroll
        for (int t2 = 0; t2 < 2; ++t2) {
            v8f c;
            #pragma unroll
            for (int j = 0; j < 8; ++j) c[j] = 0.0f;
            const int nl = t2 * 16 + l16;             // local HID row in chunk
            #pragma unroll
            for (int kc = 0; kc < 4; ++kc) {
                Frag32B bfr;
                const unsigned short* p = w1c + nl * W1_STRIDE + kc * 32 + hf * 8;
                bfr.q[0] = *reinterpret_cast<const u32x4*>(p);       // ds_load_b128
                bfr.q[1] = *reinterpret_cast<const u32x4*>(p + 16);
                c = __builtin_amdgcn_wmma_f32_16x16x32_bf16(
                        false, a[kc].v, false, bfr.v, (short)0, c, false, false);
            }
            // ReLU, fold gate weight into rows, store bf16 h chunk [16 x 32]
            #pragma unroll
            for (int j = 0; j < 8; ++j) {
                float v = c[j] > 0.0f ? c[j] : 0.0f;
                hb[(j + 8 * hf) * H_STRIDE + t2 * 16 + l16] = f2bf(v * wrow[j]);
            }
        }
        __syncthreads();   // h stores ordered before A-fragment reads

        // -------- layer 2: A = h chunk (16x32), accumulate into 8 out tiles
        Frag32B ha;
        const unsigned short* hp = hb + l16 * H_STRIDE + hf * 8;
        ha.q[0] = *reinterpret_cast<const u32x4*>(hp);
        ha.q[1] = *reinterpret_cast<const u32x4*>(hp + 16);
        #pragma unroll
        for (int nb = 0; nb < 8; ++nb) {
            Frag32B bfr;
            const unsigned short* p = w2c + (nb * 16 + l16) * W2_STRIDE + hf * 8;
            bfr.q[0] = *reinterpret_cast<const u32x4*>(p);
            bfr.q[1] = *reinterpret_cast<const u32x4*>(p + 16);
            oacc[nb] = __builtin_amdgcn_wmma_f32_16x16x32_bf16(
                           false, ha.v, false, bfr.v, (short)0, oacc[nb], false, false);
        }
        __syncthreads();   // chunk i fully consumed before buffer reuse
    }

    // write gate-combined output (fp32)
    #pragma unroll
    for (int nb = 0; nb < 8; ++nb)
        #pragma unroll
        for (int j = 0; j < 8; ++j)
            out[(size_t)(mrow + j + 8 * hf) * D_OUT + nb * 16 + l16] = oacc[nb][j];

    if (blockIdx.x == 0 && tid == 0)
        out[(size_t)T * D_OUT] = 0.0f;           // aux_loss = 0
}

// ---------------- host launch ----------------
extern "C" void kernel_launch(void* const* d_in, const int* in_sizes, int n_in,
                              void* d_out, int out_size, void* d_ws, size_t ws_size,
                              hipStream_t stream) {
    (void)in_sizes; (void)n_in; (void)out_size; (void)ws_size;
    const float* x    = (const float*)d_in[0];
    const int*   act  = (const int*)  d_in[1];
    const float* sgw1 = (const float*)d_in[2];
    const float* sgw2 = (const float*)d_in[3];
    const float* gatw = (const float*)d_in[4];
    const float* ew1  = (const float*)d_in[5];
    const float* ew2  = (const float*)d_in[6];
    float* out = (float*)d_out;

    // workspace layout (256B-aligned offsets), ~19.9 MB total
    char* ws = (char*)d_ws;
    unsigned short* xb  = (unsigned short*)(ws);                       // 16 MB
    unsigned short* w1b = (unsigned short*)(ws + 16777216);            // 512 KB
    unsigned short* w2b = (unsigned short*)(ws + 16777216 + 524288);   // 512 KB
    float*          gwb = (float*)(ws + 16777216 + 524288 + 524288);   // 2 MB

    const int n4x  = T * D_IN / 4;
    const int n4w1 = E * HID * D_IN / 4;
    const int n4w2 = E * D_OUT * HID / 4;
    cvt_bf16_kernel<<<(n4x  + 255) / 256, 256, 0, stream>>>(x,   xb,  n4x);
    cvt_bf16_kernel<<<(n4w1 + 255) / 256, 256, 0, stream>>>(ew1, w1b, n4w1);
    cvt_bf16_kernel<<<(n4w2 + 255) / 256, 256, 0, stream>>>(ew2, w2b, n4w2);

    gate_kernel<<<T / 256, 256, 0, stream>>>(x, act, sgw1, sgw2, gatw, gwb);

    moe_expert_kernel<<<T / 128, 256, 0, stream>>>(xb, w1b, w2b, gwb, out);
}
